// MultiHeadSelfAttention_11579231830224
// MI455X (gfx1250) — matrix operand order repaired and checked by reading, not compiled, not run on hardware
//
#include <hip/hip_runtime.h>

// ---------------------------------------------------------------------------
// MI455X (gfx1250) multi-head self-attention, bf16 WMMA path.
// out = softmax((xWq)(xWk)^T / 8) (xWv) Wo + bo
// v_wmma_f32_16x16x32_bf16 everywhere (wave32 fragments per ISA 7.12.2),
// fp32 accumulation, flash-style online softmax (row-sum via ones-matrix
// WMMA, row-max via v_permlane16_b32 XOR butterflies). Tiles staged into LDS
// with GLOBAL_LOAD_ASYNC_TO_LDS_B128 (ASYNCcnt) and double-buffered.
// ---------------------------------------------------------------------------

#define EMBED 1024
#define SEQ   2048
#define BATCH 4
#define HEADS 16
#define HDIM  64
#define MTOT  (BATCH * SEQ)

typedef __bf16 bf16_t;
typedef bf16_t bf16x16 __attribute__((ext_vector_type(16)));
typedef float  f32x8   __attribute__((ext_vector_type(8)));
typedef unsigned int u32x4 __attribute__((ext_vector_type(4)));
typedef int v4i __attribute__((vector_size(16)));   // matches async builtin pointee
typedef unsigned short u16;

// fp32 -> bf16 (native v_cvt, round-to-nearest-even)
__device__ __forceinline__ u16 f2bf(float f) {
  union { bf16_t b; u16 u; } cv;
  cv.b = (bf16_t)f;
  return cv.u;
}

// 32-byte WMMA operand fragment (16 x bf16 per lane)
union FragAB {
  u32x4    q[2];
  unsigned u32a[8];
  u16      us[16];
  bf16x16  v;
};

__device__ __forceinline__ f32x8 wmma_bf16(const FragAB& a, const FragAB& b, f32x8 c) {
  return __builtin_amdgcn_wmma_f32_16x16x32_bf16(
      /*neg_a=*/false, a.v, /*neg_b=*/false, b.v,
      /*c_mod=*/(short)0, c, /*reuse_a=*/false, /*reuse_b=*/false);
}

// XOR-lane exchange within each 16-lane row. Preferred path: v_permlane16_b32
// (pure VALU, no DS, no wait). SL/SH are the 16 nibble lane selectors for the
// XOR pattern; MASK is the ds_swizzle fallback pattern.
template <unsigned SL, unsigned SH, int MASK>
__device__ __forceinline__ float lane_xor16(float v) {
#if __has_builtin(__builtin_amdgcn_permlane16)
  unsigned u = __float_as_uint(v);
  return __uint_as_float(
      __builtin_amdgcn_permlane16(u, u, SL, SH, false, false));
#else
  return __int_as_float(
      __builtin_amdgcn_ds_swizzle(__float_as_int(v), (MASK << 10) | 0x1F));
#endif
}

// ---------------------------------------------------------------------------
// Async global->LDS (CDNA5 GLOBAL_LOAD_ASYNC_TO_LDS_B128, ASYNCcnt-tracked).
// ---------------------------------------------------------------------------
#if __has_builtin(__builtin_amdgcn_global_load_async_to_lds_b128)
#define HAS_ASYNC_LDS 1
#else
#define HAS_ASYNC_LDS 0
#endif

__device__ __forceinline__ void async_copy16(const u16* g, u16* lds_ptr) {
#if HAS_ASYNC_LDS
  __builtin_amdgcn_global_load_async_to_lds_b128(
      (__attribute__((address_space(1))) v4i*)g,
      (__attribute__((address_space(3))) v4i*)lds_ptr,
      /*imm offset=*/0, /*cpol=*/0);
#else
  *(u32x4*)lds_ptr = *(const u32x4*)g;
#endif
}

template <int N>
__device__ __forceinline__ void wait_async_lds() {
#if HAS_ASYNC_LDS
#if __has_builtin(__builtin_amdgcn_s_wait_asynccnt)
  __builtin_amdgcn_s_wait_asynccnt(N);
#else
  asm volatile("s_wait_asynccnt %0" ::"i"(N) : "memory");
#endif
#endif
}

// ---------------------------------------------------------------------------
// fp32 -> bf16 elementwise (x)
// ---------------------------------------------------------------------------
__global__ __launch_bounds__(256) void cvt_f32_bf16(const float* __restrict__ src,
                                                    u16* __restrict__ dst) {
  int i = (blockIdx.x * 256 + threadIdx.x) * 4;
  float4 f = *(const float4*)(src + i);
  uint2 o;
  o.x = (unsigned)f2bf(f.x) | ((unsigned)f2bf(f.y) << 16);
  o.y = (unsigned)f2bf(f.z) | ((unsigned)f2bf(f.w) << 16);
  *(uint2*)(dst + i) = o;
}

// ---------------------------------------------------------------------------
// W[K][N] fp32 -> Wt[N][K] bf16  (K = N = EMBED)
// ---------------------------------------------------------------------------
__global__ __launch_bounds__(256) void wt_transpose(const float* __restrict__ W,
                                                    u16* __restrict__ Wt) {
  int idx = blockIdx.x * 256 + threadIdx.x;     // over EMBED*EMBED
  int n = idx >> 10;                            // / EMBED
  int k = idx & (EMBED - 1);
  Wt[(size_t)n * EMBED + k] = f2bf(W[(size_t)k * EMBED + n]);
}

// ---------------------------------------------------------------------------
// V [B*S, D] bf16 -> Vt [B,H,Hd,S] bf16
// ---------------------------------------------------------------------------
__global__ __launch_bounds__(256) void v_transpose(const u16* __restrict__ V,
                                                   u16* __restrict__ Vt) {
  int idx = blockIdx.x * 256 + threadIdx.x;     // over B*H*Hd*S
  int s  = idx & (SEQ - 1);
  int r  = idx >> 11;                           // / SEQ
  int hd = r & (HDIM - 1);
  int bh = r >> 6;                              // b*HEADS + h
  int b  = bh >> 4;
  int h  = bh & (HEADS - 1);
  Vt[idx] = V[((size_t)(b * SEQ + s)) * EMBED + h * HDIM + hd];
}

// ---------------------------------------------------------------------------
// GEMM: C[M,N] = A[M,K] * Wt[N,K]^T + bias.   A,Wt bf16; C fp32 or bf16.
// Block tile 128(M) x 128(N), K-step 32, double-buffered async LDS staging.
// 8 waves: 4(M) x 2(N); each wave 32x64 = 2x4 wmma accumulators.
// Per thread per K-tile: 2 A + 2 B chunks (16B each) -> 4 async loads.
// ---------------------------------------------------------------------------
template <bool OUT_BF16>
__global__ __launch_bounds__(256)
void gemm_bf16(const u16* __restrict__ A, const u16* __restrict__ Wt,
               const float* __restrict__ bias, void* __restrict__ Cout,
               int Ndim, int Kdim) {
  __shared__ u16 sA[2][128][40];    // +8 pad keeps 16B alignment
  __shared__ u16 sB[2][128][40];

  const int tid  = threadIdx.x;
  const int lane = tid & 31;
  const int w    = tid >> 5;
  const int wm   = w >> 1;          // 0..3
  const int wn   = w & 1;           // 0..1
  const int l15  = lane & 15;
  const int lh   = lane >> 4;       // half-wave (ISA fragment layout selector)
  const int mBase = blockIdx.y * 128;
  const int nBase = blockIdx.x * 128;

  // 128 rows x 4 chunks = 512 chunks each for A and B; 2 per thread (rows +64)
  const int ldRow = tid >> 2, ldCol = (tid & 3) * 8;
  const u16* aG = &A[(size_t)(mBase + ldRow) * Kdim + ldCol];
  const u16* bG = &Wt[(size_t)(nBase + ldRow) * Kdim + ldCol];
  const size_t rowOff64 = (size_t)64 * Kdim;

  auto issue = [&](int kt, int buf) {
    const int ko = kt * 32;
    async_copy16(aG + ko,            &sA[buf][ldRow][ldCol]);
    async_copy16(aG + ko + rowOff64, &sA[buf][ldRow + 64][ldCol]);
    async_copy16(bG + ko,            &sB[buf][ldRow][ldCol]);
    async_copy16(bG + ko + rowOff64, &sB[buf][ldRow + 64][ldCol]);
  };

  f32x8 acc[2][4] = {};

  auto compute = [&](int cur) {
    FragAB af[2], bf[4];
#pragma unroll
    for (int i = 0; i < 2; i++) {          // A frag: row m=l15, kb=lh*8
      int row = wm * 32 + i * 16 + l15;
      int kb  = lh * 8;
      af[i].q[0] = *(const u32x4*)&sA[cur][row][kb];
      af[i].q[1] = *(const u32x4*)&sA[cur][row][kb + 16];
    }
#pragma unroll
    for (int j = 0; j < 4; j++) {          // B frag: col n=l15, kb=lh*8
      int row = wn * 64 + j * 16 + l15;
      int ko  = lh * 8;
      bf[j].q[0] = *(const u32x4*)&sB[cur][row][ko];
      bf[j].q[1] = *(const u32x4*)&sB[cur][row][ko + 16];
    }
#pragma unroll
    for (int i = 0; i < 2; i++)
#pragma unroll
      for (int j = 0; j < 4; j++)
        acc[i][j] = wmma_bf16(af[i], bf[j], acc[i][j]);
  };

  const int NT = Kdim / 32;
  issue(0, 0);
  for (int kt = 0; kt < NT - 1; ++kt) {
    issue(kt + 1, (kt + 1) & 1);
    wait_async_lds<4>();
    __syncthreads();
    compute(kt & 1);
    __syncthreads();                       // readers done before buf reuse
  }
  wait_async_lds<0>();
  __syncthreads();
  compute((NT - 1) & 1);

  // C/D layout: vgpr r -> M = r + lh*8 ; N = l15
#pragma unroll
  for (int i = 0; i < 2; i++) {
    int mrowBase = mBase + wm * 32 + i * 16 + lh * 8;
#pragma unroll
    for (int j = 0; j < 4; j++) {
      int n = nBase + wn * 64 + j * 16 + l15;
      float bv = bias[n];
#pragma unroll
      for (int r = 0; r < 8; r++) {
        float val = acc[i][j][r] + bv;
        size_t off = (size_t)(mrowBase + r) * Ndim + n;
        if (OUT_BF16) ((u16*)Cout)[off]   = f2bf(val);
        else          ((float*)Cout)[off] = val;
      }
    }
  }
}

// ---------------------------------------------------------------------------
// Flash attention. Block = 128 threads (4 waves). Wave handles 16 queries,
// full Hd=64. Per 32-key step: K (32x64) and Vt (64x32) tiles async-loaded
// into double-buffered LDS; QK^T = 4 wmma; row-max via 4 permlane16 XOR
// butterfly steps (8 rows batched per step); P written to LDS as bf16 and
// read back as packed u32 A-fragments; row-sum = P x ones (1 wmma); PV = 4.
// ---------------------------------------------------------------------------
__global__ __launch_bounds__(128)
void flash_attn(const u16* __restrict__ Q, const u16* __restrict__ Kmat,
                const u16* __restrict__ Vt, u16* __restrict__ O) {
  __shared__ u16 sK[2][32][72];     // 32 keys x 64 hd (+8 pad)
  __shared__ u16 sV[2][64][40];     // 64 hd x 32 keys (+8 pad)
  __shared__ u16 pS16[4][16][36];   // per-wave P tile, bf16 (16 q x 32 k, +4 pad)

  const int tid  = threadIdx.x;
  const int lane = tid & 31;
  const int w    = tid >> 5;
  const int l15  = lane & 15;
  const int lh   = lane >> 4;
  const int b    = blockIdx.z;
  const int h    = blockIdx.y;
  const int qbase = blockIdx.x * 64 + w * 16;

  const size_t rowOff = (size_t)(b * SEQ) * EMBED + h * HDIM;
  const u16* Qp = Q    + rowOff;
  const u16* Kp = Kmat + rowOff;
  const u16* Vh = Vt + (size_t)((b * HEADS + h) * HDIM) * SEQ;

  // K tile: 32 rows x 8 chunks = 256; V tile: 64 rows x 4 chunks = 256
  const int kRow0 = tid >> 3,         kCol0 = (tid & 7) * 8;
  const int kRow1 = (tid + 128) >> 3, kCol1 = kCol0;
  const int vRow0 = tid >> 2,         vCol0 = (tid & 3) * 8;
  const int vRow1 = (tid + 128) >> 2, vCol1 = vCol0;

  auto issue = [&](int jt, int buf) {
    const int j = jt * 32;
    async_copy16(Kp + (size_t)(j + kRow0) * EMBED + kCol0, &sK[buf][kRow0][kCol0]);
    async_copy16(Kp + (size_t)(j + kRow1) * EMBED + kCol1, &sK[buf][kRow1][kCol1]);
    async_copy16(Vh + (size_t)vRow0 * SEQ + j + vCol0,     &sV[buf][vRow0][vCol0]);
    async_copy16(Vh + (size_t)vRow1 * SEQ + j + vCol1,     &sV[buf][vRow1][vCol1]);
  };

  // Q A-fragments for the two 32-wide K (=Hd) steps (loaded once, from global)
  FragAB qf[2];
  {
    int m  = qbase + l15;
    int kb = lh * 8;
#pragma unroll
    for (int ks = 0; ks < 2; ks++) {
      const u16* p = Qp + (size_t)m * EMBED + ks * 32;
      qf[ks].q[0] = *(const u32x4*)(p + kb);
      qf[ks].q[1] = *(const u32x4*)(p + kb + 16);
    }
  }

  FragAB onesB;                       // 32x16 all-ones B matrix for row sums
#pragma unroll
  for (int i = 0; i < 16; i++) onesB.us[i] = 0x3F80;   // bf16 1.0

  f32x8 acc[4] = {};
  float mrow[8], lrow[8];
#pragma unroll
  for (int r = 0; r < 8; r++) { mrow[r] = -1e30f; lrow[r] = 0.f; }

  auto compute = [&](int cur) {
    // S = Q K^T : B frag col n = key (row l15 / 16+l15 of sK), k = hd
    f32x8 s0 = {}, s1 = {};
#pragma unroll
    for (int ks = 0; ks < 2; ks++) {
      int koff = ks * 32 + lh * 8;
      FragAB k0f, k1f;
      k0f.q[0] = *(const u32x4*)&sK[cur][l15][koff];
      k0f.q[1] = *(const u32x4*)&sK[cur][l15][koff + 16];
      k1f.q[0] = *(const u32x4*)&sK[cur][16 + l15][koff];
      k1f.q[1] = *(const u32x4*)&sK[cur][16 + l15][koff + 16];
      s0 = wmma_bf16(qf[ks], k0f, s0);
      s1 = wmma_bf16(qf[ks], k1f, s1);
    }

    const float scale = 0.125f;                   // 1/sqrt(64)

    // Row max: 8 independent butterflies, 4 XOR steps via v_permlane16_b32
    float mx[8];
#pragma unroll
    for (int r = 0; r < 8; r++) mx[r] = fmaxf(s0[r], s1[r]);
#define BFLY_STEP(SL, SH, MASK)                                          \
    {                                                                    \
      float t0[8];                                                       \
      _Pragma("unroll")                                                  \
      for (int r = 0; r < 8; r++) t0[r] = lane_xor16<SL, SH, MASK>(mx[r]); \
      _Pragma("unroll")                                                  \
      for (int r = 0; r < 8; r++) mx[r] = fmaxf(mx[r], t0[r]);           \
    }
    BFLY_STEP(0x67452301u, 0xEFCDAB89u, 1)   // xor 1
    BFLY_STEP(0x54761032u, 0xDCFE98BAu, 2)   // xor 2
    BFLY_STEP(0x32107654u, 0xBA98FEDCu, 4)   // xor 4
    BFLY_STEP(0xFEDCBA98u, 0x76543210u, 8)   // xor 8
#undef BFLY_STEP

    float corr[8];
#pragma unroll
    for (int r = 0; r < 8; r++) {
      float mnew = fmaxf(mrow[r], mx[r] * scale);
      corr[r] = __expf(mrow[r] - mnew);
      mrow[r] = mnew;
      float p0 = __expf(s0[r] * scale - mnew);
      float p1 = __expf(s1[r] * scale - mnew);
      int m = r + lh * 8;                         // C/D layout -> LDS P tile
      pS16[w][m][l15]      = f2bf(p0);
      pS16[w][m][16 + l15] = f2bf(p1);
    }
    asm volatile("s_wait_dscnt 0x0" ::: "memory");

#pragma unroll
    for (int t = 0; t < 4; t++)
#pragma unroll
      for (int r = 0; r < 8; r++)
        acc[t][r] *= corr[r];

    // Repack P into A-fragment layout: lane m=l15, kb=lh*8. Adjacent k pairs
    // are adjacent bf16 in LDS -> one u32 load per fragment VGPR.
    FragAB pf;
    {
      int m = l15, kb = lh * 8;
#pragma unroll
      for (int v2 = 0; v2 < 4; v2++) {
        pf.u32a[v2]     = *(const unsigned*)&pS16[w][m][kb + 2 * v2];
        pf.u32a[4 + v2] = *(const unsigned*)&pS16[w][m][kb + 16 + 2 * v2];
      }
    }
    asm volatile("s_wait_dscnt 0x0" ::: "memory");

    // Row sum via WMMA: P(16x32) x ones(32x16) -> every column = row sum,
    // D fragment layout matches lrow indexing (row m = r + lh*8).
    f32x8 zero = {};
    f32x8 rs = wmma_bf16(pf, onesB, zero);
#pragma unroll
    for (int r = 0; r < 8; r++)
      lrow[r] = lrow[r] * corr[r] + rs[r];

#pragma unroll
    for (int t = 0; t < 4; t++) {                 // PV: B frag from sV
      FragAB vf;
      vf.q[0] = *(const u32x4*)&sV[cur][t * 16 + l15][lh * 8];
      vf.q[1] = *(const u32x4*)&sV[cur][t * 16 + l15][lh * 8 + 16];
      acc[t] = wmma_bf16(pf, vf, acc[t]);
    }
  };

  const int NT = SEQ / 32;
  issue(0, 0);
  for (int jt = 0; jt < NT - 1; ++jt) {
    issue(jt + 1, (jt + 1) & 1);
    wait_async_lds<4>();
    __syncthreads();
    compute(jt & 1);
    __syncthreads();                              // readers done before buf reuse
  }
  wait_async_lds<0>();
  __syncthreads();
  compute((NT - 1) & 1);

#pragma unroll
  for (int t = 0; t < 4; t++) {
#pragma unroll
    for (int r = 0; r < 8; r++) {
      int token = qbase + r + lh * 8;
      float ov = acc[t][r] / lrow[r];
      O[(size_t)(b * SEQ + token) * EMBED + h * HDIM + t * 16 + l15] = f2bf(ov);
    }
  }
}

// ---------------------------------------------------------------------------
// Host-side orchestration
// ---------------------------------------------------------------------------
extern "C" void kernel_launch(void* const* d_in, const int* in_sizes, int n_in,
                              void* d_out, int out_size, void* d_ws, size_t ws_size,
                              hipStream_t stream) {
  (void)in_sizes; (void)n_in; (void)out_size; (void)ws_size;
  const float* x  = (const float*)d_in[0];
  const float* Wq = (const float*)d_in[1];
  const float* bq = (const float*)d_in[2];
  const float* Wk = (const float*)d_in[3];
  const float* bk = (const float*)d_in[4];
  const float* Wv = (const float*)d_in[5];
  const float* bv = (const float*)d_in[6];
  const float* Wo = (const float*)d_in[7];
  const float* bo = (const float*)d_in[8];
  float* out = (float*)d_out;

  char* ws = (char*)d_ws;
  const size_t actBytes = (size_t)MTOT * EMBED * sizeof(u16);   // 16 MB
  const size_t wBytes   = (size_t)EMBED * EMBED * sizeof(u16);  // 2 MB
  u16* xb  = (u16*)ws;  ws += actBytes;
  u16* Wqt = (u16*)ws;  ws += wBytes;
  u16* Wkt = (u16*)ws;  ws += wBytes;
  u16* Wvt = (u16*)ws;  ws += wBytes;
  u16* Wot = (u16*)ws;  ws += wBytes;
  u16* Qb  = (u16*)ws;  ws += actBytes;
  u16* Kb  = (u16*)ws;  ws += actBytes;
  u16* Vb  = (u16*)ws;  ws += actBytes;
  u16* Vtp = (u16*)ws;  ws += actBytes;
  u16* Ob  = (u16*)ws;  ws += actBytes;

  cvt_f32_bf16<<<(MTOT * EMBED) / (256 * 4), 256, 0, stream>>>(x, xb);
  wt_transpose<<<(EMBED * EMBED) / 256, 256, 0, stream>>>(Wq, Wqt);
  wt_transpose<<<(EMBED * EMBED) / 256, 256, 0, stream>>>(Wk, Wkt);
  wt_transpose<<<(EMBED * EMBED) / 256, 256, 0, stream>>>(Wv, Wvt);
  wt_transpose<<<(EMBED * EMBED) / 256, 256, 0, stream>>>(Wo, Wot);

  dim3 gg(EMBED / 128, MTOT / 128);
  gemm_bf16<true><<<gg, 256, 0, stream>>>(xb, Wqt, bq, Qb, EMBED, EMBED);
  gemm_bf16<true><<<gg, 256, 0, stream>>>(xb, Wkt, bk, Kb, EMBED, EMBED);
  gemm_bf16<true><<<gg, 256, 0, stream>>>(xb, Wvt, bv, Vb, EMBED, EMBED);

  v_transpose<<<(BATCH * HEADS * HDIM * SEQ) / 256, 256, 0, stream>>>(Vb, Vtp);

  flash_attn<<<dim3(SEQ / 64, HEADS, BATCH), 128, 0, stream>>>(Qb, Kb, Vtp, Ob);

  gemm_bf16<false><<<gg, 256, 0, stream>>>(Ob, Wot, bo, out, EMBED, EMBED);
}
